// AdaFS_hard_75342316306484
// MI455X (gfx1250) — matrix operand order repaired and checked by reading, not compiled, not run on hardware
//
#include <hip/hip_runtime.h>
#include <hip/hip_bf16.h>

typedef __attribute__((ext_vector_type(16))) __bf16 v16bf;
typedef __attribute__((ext_vector_type(8)))  __bf16 v8bf;
typedef __attribute__((ext_vector_type(8)))  float  v8f;

constexpr int B  = 16384;
constexpr int F  = 5;
constexpr int E  = 2560;   // D*F
constexpr int EH = 1280;   // E/2
constexpr int ROWS = 32;   // M-tile per block (2 WMMA M-tiles -> 2x B reuse)
constexpr int ASTRIDE = E + 8;  // 2568 bf16 = 5136B row stride: 16B aligned, bank-staggered
constexpr int CHUNK = 256;      // f32 elements per row per async staging chunk

// ---------------------------------------------------------------------------
// Transpose K x N f32  ->  N x K bf16  (makes WMMA B-fragments contiguous)
// ---------------------------------------------------------------------------
__global__ void transpose_to_bf16(const float* __restrict__ in,
                                  __bf16* __restrict__ out, int N, int K) {
  int idx = blockIdx.x * blockDim.x + threadIdx.x;
  if (idx >= N * K) return;
  int n = idx / K, k = idx - n * K;
  out[idx] = (__bf16)in[(size_t)k * N + n];
}

// ---------------------------------------------------------------------------
// Controller: h1 = relu(flat@cW1+cb1) via WMMA, then 64->32->5, softmax,
// top-3 (drop two smallest, latest-index tie preference), renormalize -> mask
// ---------------------------------------------------------------------------
__global__ __launch_bounds__(128) void controller_kernel(
    const float* __restrict__ flat, const __bf16* __restrict__ cW1t,
    const float* __restrict__ cb1, const float* __restrict__ cW2,
    const float* __restrict__ cb2, const float* __restrict__ cW3,
    const float* __restrict__ cb3, float* __restrict__ mask) {
  __shared__ float h1[16][72];
  __shared__ float h2[16][33];

  const int tid  = threadIdx.x;
  const int wave = tid >> 5, lane = tid & 31;
  const int half = lane >> 4, m = lane & 15;
  const int row0 = blockIdx.x * 16;
  const int nbase = wave * 16;

  v8f c = {};
  const float*  arow = flat  + (size_t)(row0 + m) * E;
  const __bf16* brow = cW1t + (size_t)(nbase + m) * E;
  for (int k0 = 0; k0 < E; k0 += 32) {
    const float* ap = arow + k0 + half * 16;
    v16bf a;
#pragma unroll
    for (int j = 0; j < 8; ++j) {
      a[j]     = (__bf16)ap[j];        // VGPR0-3: K = k0 + half*16 + 0..7
      a[8 + j] = (__bf16)ap[8 + j];    // VGPR4-7: K = k0 + half*16 + 8..15
    }
    v16bf b = *(const v16bf*)(brow + k0 + half * 16);
    c = __builtin_amdgcn_wmma_f32_16x16x32_bf16(false, a, false, b,
                                                (short)0, c, false, false);
  }
  {
    const int col = nbase + m;
    const float bias = cb1[col];
#pragma unroll
    for (int i = 0; i < 8; ++i) {
      float v = c[i] + bias;
      h1[i + half * 8][col] = v > 0.f ? v : 0.f;
    }
  }
  __syncthreads();

#pragma unroll
  for (int t = 0; t < 4; ++t) {
    int idx = tid * 4 + t;
    int r = idx >> 5, cc = idx & 31;
    float s = cb2[cc];
    for (int k = 0; k < 64; ++k) s += h1[r][k] * cW2[k * 32 + cc];
    h2[r][cc] = s > 0.f ? s : 0.f;
  }
  __syncthreads();

  if (tid < 16) {
    int r = tid;
    float wgt[5];
    float mx = -1e30f;
#pragma unroll
    for (int f = 0; f < F; ++f) {
      float s = cb3[f];
      for (int k = 0; k < 32; ++k) s += h2[r][k] * cW3[k * F + f];
      wgt[f] = s;
      mx = s > mx ? s : mx;
    }
    float sum = 0.f;
#pragma unroll
    for (int f = 0; f < F; ++f) { wgt[f] = __expf(wgt[f] - mx); sum += wgt[f]; }
    float inv = 1.f / sum;
#pragma unroll
    for (int f = 0; f < F; ++f) wgt[f] *= inv;
    int i1 = 0;
#pragma unroll
    for (int f = 1; f < F; ++f) if (wgt[f] <= wgt[i1]) i1 = f;      // last argmin
    int i2 = (i1 == 0) ? 1 : 0;
#pragma unroll
    for (int f = 0; f < F; ++f)
      if (f != i1 && wgt[f] <= wgt[i2]) i2 = f;                     // last argmin of rest
    float s2 = 0.f;
#pragma unroll
    for (int f = 0; f < F; ++f) if (f != i1 && f != i2) s2 += wgt[f];
    float inv2 = 1.f / s2;
#pragma unroll
    for (int f = 0; f < F; ++f)
      mask[(size_t)(row0 + r) * F + f] = (f != i1 && f != i2) ? wgt[f] * inv2 : 0.f;
  }
}

// ---------------------------------------------------------------------------
// Main MLP: y1 = relu(gated@mW1+mb1) via WMMA (2 M-tiles per B fragment),
// fused into y2 via shuffle-reduced epilogue, then out = relu(y2)@oW + ob.
// Field rows staged via global_load_async_to_lds_b128 (ASYNCcnt), then
// gated + converted to bf16 in LDS.
// ---------------------------------------------------------------------------
__global__ __launch_bounds__(256) void main_mlp_kernel(
    const float* __restrict__ field, const float* __restrict__ mask,
    const __bf16* __restrict__ mW1t, const float* __restrict__ mb1,
    const float* __restrict__ mW2, const float* __restrict__ mb2,
    const float* __restrict__ oW, const float* __restrict__ ob,
    float* __restrict__ out) {
  __shared__ __bf16 As[ROWS * ASTRIDE];     // 160.5 KB gated bf16 A panel
  __shared__ float  raw[ROWS * CHUNK];      // 32 KB async staging buffer
  __shared__ float  y2s[ROWS][5];
  __shared__ float  mrow[ROWS][5];

  const int tid  = threadIdx.x;
  const int row0 = blockIdx.x * ROWS;

  if (tid < ROWS * F) {
    int r = tid / 5, g = tid - r * 5;
    mrow[r][g] = mask[(size_t)(row0 + r) * F + g];
    y2s[r][g]  = mb2[g];
  }
  __syncthreads();

  // --- Stage gated rows: async copy raw f32 chunk -> LDS, then gate+cvt ---
  {
    const float* fbase = field + (size_t)row0 * E;
    const unsigned rawBase = (unsigned)(size_t)(void*)&raw[0];
    for (int cch = 0; cch < E / CHUNK; ++cch) {
#pragma unroll
      for (int q = 0; q < (ROWS * CHUNK / 4) / 256; ++q) {
        int i = q * 256 + tid;                  // b128 transfer index
        int r = i >> 6, seg = i & 63;           // 64 x b128 per row-chunk
        unsigned ldsDst = rawBase + ((unsigned)i << 4);
        unsigned gOff = (unsigned)((r * E + cch * CHUNK + (seg << 2)) << 2);
        asm volatile("global_load_async_to_lds_b128 %0, %1, %2 offset:0"
                     :
                     : "v"(ldsDst), "v"(gOff), "s"(fbase)
                     : "memory");
      }
      asm volatile("s_wait_asynccnt 0" ::: "memory");
      __syncthreads();
#pragma unroll
      for (int q = 0; q < (ROWS * CHUNK) / 256; ++q) {
        int i = q * 256 + tid;
        int r = i >> 8, e0 = i & (CHUNK - 1);
        int e = cch * CHUNK + e0;
        As[r * ASTRIDE + e] = (__bf16)(raw[r * CHUNK + e0] * mrow[r][e % 5]);
      }
      __syncthreads();
    }
  }

  // --- WMMA sweep: each wave owns n-tiles, 2 M-tiles share each B frag ---
  const int wave = tid >> 5, lane = tid & 31;
  const int half = lane >> 4, m = lane & 15;

  for (int nt = wave; nt < EH / 16; nt += 8) {
    const int nbase = nt * 16;
    const int col   = nbase + m;
    const __bf16* brow = mW1t + (size_t)col * E;
    const __bf16* a0 = As + m * ASTRIDE;          // rows 0..15
    const __bf16* a1 = As + (16 + m) * ASTRIDE;   // rows 16..31
    v8f c0 = {}, c1 = {};
#pragma unroll 4
    for (int k0 = 0; k0 < E; k0 += 32) {
      v16bf b = *(const v16bf*)(brow + k0 + half * 16);      // 2x b128
      __builtin_prefetch((const void*)(brow + k0 + 256), 0, 0);
      v8bf lo0 = *(const v8bf*)(a0 + k0 + half * 16);        // ds_load_b128
      v8bf hi0 = *(const v8bf*)(a0 + k0 + half * 16 + 8);
      v16bf A0 = __builtin_shufflevector(lo0, hi0,
                   0, 1, 2, 3, 4, 5, 6, 7, 8, 9, 10, 11, 12, 13, 14, 15);
      c0 = __builtin_amdgcn_wmma_f32_16x16x32_bf16(false, A0, false, b,
                                                   (short)0, c0, false, false);
      v8bf lo1 = *(const v8bf*)(a1 + k0 + half * 16);
      v8bf hi1 = *(const v8bf*)(a1 + k0 + half * 16 + 8);
      v16bf A1 = __builtin_shufflevector(lo1, hi1,
                   0, 1, 2, 3, 4, 5, 6, 7, 8, 9, 10, 11, 12, 13, 14, 15);
      c1 = __builtin_amdgcn_wmma_f32_16x16x32_bf16(false, A1, false, b,
                                                   (short)0, c1, false, false);
    }
    // Fused epilogue: relu(y1) * mW2 scaled per-lane (per-column), then
    // cross-column sum via 4 shfl_xor steps within each 16-lane half.
    // Lanes 0/16 commit one LDS atomic per (row, gate): 16x fewer atomics.
    const float bias = mb1[col];
    float w2[5];
#pragma unroll
    for (int g = 0; g < F; ++g) w2[g] = mW2[col * F + g];
#pragma unroll
    for (int i = 0; i < 8; ++i) {
      float v0 = c0[i] + bias;
      v0 = v0 > 0.f ? v0 : 0.f;
      float v1 = c1[i] + bias;
      v1 = v1 > 0.f ? v1 : 0.f;
#pragma unroll
      for (int g = 0; g < F; ++g) {
        float p0 = v0 * w2[g];
        float p1 = v1 * w2[g];
#pragma unroll
        for (int s = 1; s < 16; s <<= 1) {
          p0 += __shfl_xor(p0, s, 32);
          p1 += __shfl_xor(p1, s, 32);
        }
        if (m == 0) {                       // lane 0 -> rows i / lane 16 -> rows 8+i
          int rm = i + half * 8;
          atomicAdd(&y2s[rm][g],      p0);
          atomicAdd(&y2s[16 + rm][g], p1);
        }
      }
    }
  }
  __syncthreads();

  if (tid < ROWS) {
    float acc = ob[0];
#pragma unroll
    for (int g = 0; g < F; ++g) {
      float v = y2s[tid][g];
      v = v > 0.f ? v : 0.f;
      acc += v * oW[g];
    }
    out[row0 + tid] = acc;
  }
}

// ---------------------------------------------------------------------------
extern "C" void kernel_launch(void* const* d_in, const int* in_sizes, int n_in,
                              void* d_out, int out_size, void* d_ws, size_t ws_size,
                              hipStream_t stream) {
  const float* field = (const float*)d_in[0];
  const float* cW1   = (const float*)d_in[1];
  const float* cb1   = (const float*)d_in[2];
  const float* cW2   = (const float*)d_in[3];
  const float* cb2   = (const float*)d_in[4];
  const float* cW3   = (const float*)d_in[5];
  const float* cb3   = (const float*)d_in[6];
  const float* mW1   = (const float*)d_in[7];
  const float* mb1   = (const float*)d_in[8];
  const float* mW2   = (const float*)d_in[9];
  const float* mb2   = (const float*)d_in[10];
  const float* oW    = (const float*)d_in[11];
  const float* ob    = (const float*)d_in[12];
  float* out = (float*)d_out;

  // Workspace layout (~7.2 MB): mask | cW1t bf16 | mW1t bf16
  char* w = (char*)d_ws;
  float*  mask = (float*)w;
  __bf16* cW1t = (__bf16*)(w + (size_t)B * F * sizeof(float));
  __bf16* mW1t = (__bf16*)(w + (size_t)B * F * sizeof(float) + (size_t)64 * E * 2);

  transpose_to_bf16<<<(64 * E + 255) / 256, 256, 0, stream>>>(cW1, cW1t, 64, E);
  transpose_to_bf16<<<(EH * E + 255) / 256, 256, 0, stream>>>(mW1, mW1t, EH, E);
  controller_kernel<<<B / 16, 128, 0, stream>>>(field, cW1t, cb1, cW2, cb2,
                                                cW3, cb3, mask);
  main_mlp_kernel<<<B / ROWS, 256, 0, stream>>>(field, mask, mW1t, mb1, mW2,
                                                mb2, oW, ob, out);
}
